// AttentionLayer_59828894433713
// MI455X (gfx1250) — matrix operand order repaired and checked by reading, not compiled, not run on hardware
//
#include <hip/hip_runtime.h>
#include <hip/hip_bf16.h>
#include <cstdint>

// ---------------------------------------------------------------------------
// Model constants
// ---------------------------------------------------------------------------
#define BATCH     2
#define SEQ       2048
#define DMODEL    2048
#define NHEADS    16
#define DHEAD     128
#define DFFN      8192
#define NTOK      (BATCH*SEQ)          // 4096
#define QKV_LD    (3*DMODEL)           // 6144
#define RMS_EPS   1.1920929e-07f
#define INV_SQRT_D 0.08838834764831845f  // 1/sqrt(128)

typedef __attribute__((ext_vector_type(16))) __bf16 v16bf;
typedef __attribute__((ext_vector_type(8)))  float  v8f;
typedef __attribute__((ext_vector_type(4)))  unsigned int u32x4;
typedef __attribute__((ext_vector_type(8)))  int i32x8;
typedef __attribute__((ext_vector_type(4)))  int i32x4;

#if defined(__gfx1250__) && __has_builtin(__builtin_amdgcn_tensor_load_to_lds) && \
    __has_builtin(__builtin_amdgcn_s_wait_tensorcnt)
#define USE_TDM 1
#else
#define USE_TDM 0
#endif

union Frag16 { v16bf v; uint4 q[2]; };

// A-fragment (16x32 bf16): lane row handled by caller; per-lane K pattern:
//   elems 0..7  = k0 + h*8 .. +7, elems 8..15 = k0 + 16 + h*8 ..
__device__ __forceinline__ v16bf load_frag_a(const __bf16* rowp, int h) {
    Frag16 f;
    f.q[0] = *(const uint4*)(rowp + h * 8);
    f.q[1] = *(const uint4*)(rowp + 16 + h * 8);
    return f.v;
}
// B-fragment (32x16 bf16) from N-major storage: per-lane K pattern same as A:
//   elems 0..7 = k0 + h*8 .., elems 8..15 = k0 + 16 + h*8 ..
__device__ __forceinline__ v16bf load_frag_b(const __bf16* colp, int h) {
    Frag16 f;
    f.q[0] = *(const uint4*)(colp + h * 8);
    f.q[1] = *(const uint4*)(colp + 16 + h * 8);
    return f.v;
}

__device__ __forceinline__ uint32_t lds_off(const void* p) {
    // low 32 bits of an LDS flat address are the workgroup-relative byte offset
    return (uint32_t)(uintptr_t)p;
}

// ---------------------------------------------------------------------------
// TDM: 2D tile (d0 elems/row x d1 rows of bf16) global -> LDS with row padding.
// pad_int: pad after 2^(pad_int+1) dwords; pad_amt: (pad_amt+1) dwords pad.
// Descriptor packing per CDNA5 ISA.
// This toolchain uses the 6-arg builtin: (g0, g1, g2, g3, g4, cpol).
// ---------------------------------------------------------------------------
__device__ __forceinline__ void tdm_load_2d(const void* gsrc, uint32_t ldsbyte,
                                            uint32_t d0, uint32_t d1,
                                            uint64_t stride_elems,
                                            uint32_t pad_int, uint32_t pad_amt) {
#if USE_TDM
    const uint64_t ga = (uint64_t)(uintptr_t)gsrc;
    u32x4 g0;
    g0[0] = 1u;                                   // count=1 (valid user descriptor)
    g0[1] = ldsbyte;                              // lds_addr
    g0[2] = (uint32_t)ga;                         // global_addr[31:0]
    g0[3] = (uint32_t)((ga >> 32) & 0x1FFFFFFu) | (2u << 30); // ga[56:32], type=2
    const uint64_t sb = stride_elems;             // in data_size units
    i32x8 g1;
    g1[0] = (int)((1u << 16)                      // data_size = 2 bytes
                | (1u << 20)                      // pad_enable
                | (pad_int << 22) | (pad_amt << 25));
    g1[1] = (int)((d0 & 0xFFFFu) << 16);                          // tensor_dim0 lo
    g1[2] = (int)(((d0 >> 16) & 0xFFFFu) | ((d1 & 0xFFFFu) << 16)); // td0 hi|td1 lo
    g1[3] = (int)(((d1 >> 16) & 0xFFFFu) | ((d0 & 0xFFFFu) << 16)); // td1 hi|tile_dim0
    g1[4] = (int)(d1 & 0xFFFFu);                                   // tile_dim1, tile_dim2=0
    g1[5] = (int)(uint32_t)(sb & 0xFFFFFFFFu);                     // dim0_stride lo
    g1[6] = (int)(uint32_t)((sb >> 32) & 0xFFFFu);                 // dim0_stride hi
    g1[7] = 0;
    const i32x4 z4 = {0, 0, 0, 0};
    const i32x8 z8 = {0, 0, 0, 0, 0, 0, 0, 0};
    __builtin_amdgcn_tensor_load_to_lds(g0, g1, z4, z4, z8, 0);
#else
    (void)gsrc; (void)ldsbyte; (void)d0; (void)d1;
    (void)stride_elems; (void)pad_int; (void)pad_amt;
#endif
}

// ---------------------------------------------------------------------------
// Weight cast + transpose: W (KxN f32, row-major) -> Wt (NxK bf16, row-major)
// ---------------------------------------------------------------------------
__global__ __launch_bounds__(256) void transpose_cast_kernel(
    const float* __restrict__ W, __bf16* __restrict__ Wt, int K, int N) {
    __shared__ float sh[32][33];
    const int n0 = blockIdx.x * 32, k0 = blockIdx.y * 32;
    const int c = threadIdx.x & 31, r = threadIdx.x >> 5;   // 32 cols x 8 rows
#pragma unroll
    for (int i = 0; i < 32; i += 8)
        sh[r + i][c] = W[(size_t)(k0 + r + i) * N + n0 + c];
    __syncthreads();
#pragma unroll
    for (int i = 0; i < 32; i += 8)
        Wt[(size_t)(n0 + r + i) * K + k0 + c] = (__bf16)sh[c][r + i];
}

// ---------------------------------------------------------------------------
// RMSNorm -> bf16 activations
// ---------------------------------------------------------------------------
__global__ __launch_bounds__(256) void rmsnorm_kernel(const float* __restrict__ x,
                                                      const float* __restrict__ w,
                                                      __bf16* __restrict__ xn) {
    __shared__ float red[256];
    const int row = blockIdx.x, tid = threadIdx.x;
    const float* xr = x + (size_t)row * DMODEL;
    float ss = 0.f;
    for (int i = tid; i < DMODEL; i += 256) { float v = xr[i]; ss += v * v; }
    red[tid] = ss; __syncthreads();
    for (int s = 128; s > 0; s >>= 1) {
        if (tid < s) red[tid] += red[tid + s];
        __syncthreads();
    }
    const float rs = rsqrtf(red[0] * (1.0f / DMODEL) + RMS_EPS);
    for (int i = tid; i < DMODEL; i += 256)
        xn[(size_t)row * DMODEL + i] = (__bf16)(xr[i] * rs * w[i]);
}

// ---------------------------------------------------------------------------
// Tiled bf16 WMMA GEMM: C(MxN) = A(MxK) @ Wt(NxK)^T + bias
// block = 256 (8 waves); tile 128x128; wave tile 32x64 (2x4 WMMA frags)
// A and Wt tiles are both 128 rows x 32 elems -> TDM double-buffered staging.
// MODE: 0=store f32, 1=store bf16, 2=sigmoid->f32, 3=accumulate into f32 C
// ---------------------------------------------------------------------------
#define ALD 40   // LDS row stride elems (64B data + 16B pad -> all frags 16B aligned)

__device__ __forceinline__ void gemm_tile_compute(const __bf16* Ab, const __bf16* Bb,
                                                  int wm, int wn, int ln, int hf,
                                                  v8f (&acc)[2][4]) {
    v16bf af[2], bfr[4];
#pragma unroll
    for (int i = 0; i < 2; ++i)
        af[i] = load_frag_a(Ab + (wm * 32 + i * 16 + ln) * ALD, hf);
#pragma unroll
    for (int j = 0; j < 4; ++j)
        bfr[j] = load_frag_b(Bb + (wn * 64 + j * 16 + ln) * ALD, hf);
#pragma unroll
    for (int i = 0; i < 2; ++i)
#pragma unroll
        for (int j = 0; j < 4; ++j)
            acc[i][j] = __builtin_amdgcn_wmma_f32_16x16x32_bf16(
                false, af[i], false, bfr[j], (short)0, acc[i][j], false, false);
}

template <int MODE>
__global__ __launch_bounds__(256) void gemm_bf16_kernel(
    const __bf16* __restrict__ A, const __bf16* __restrict__ Wt,
    const float* __restrict__ bias, void* __restrict__ Cout,
    int M, int N, int K)
{
    __shared__ __bf16 Ash[2][128 * ALD];
    __shared__ __bf16 Bsh[2][128 * ALD];

    const int tid = threadIdx.x;
    const int m0 = blockIdx.y * 128, n0 = blockIdx.x * 128;
    const int w = tid >> 5, lane = tid & 31, ln = lane & 15, hf = lane >> 4;
    const int wm = w & 3, wn = w >> 2;
    const int nk = K >> 5;

    v8f acc[2][4];
#pragma unroll
    for (int i = 0; i < 2; ++i)
#pragma unroll
        for (int j = 0; j < 4; ++j) acc[i][j] = (v8f){};

#if USE_TDM
    const bool issuer = (tid < 32);               // wave 0 drives the TDM
    if (issuer) {
        tdm_load_2d(A  + (size_t)m0 * K, lds_off(Ash[0]), 32, 128, (uint64_t)K, 3, 3);
        tdm_load_2d(Wt + (size_t)n0 * K, lds_off(Bsh[0]), 32, 128, (uint64_t)K, 3, 3);
    }
    for (int t = 0; t < nk; ++t) {
        if (issuer) __builtin_amdgcn_s_wait_tensorcnt(0);
        __syncthreads();                          // publish TDM data; retire buf t-1
        const int cur = t & 1;
        if (issuer && (t + 1) < nk) {
            const int k0n = (t + 1) << 5;
            tdm_load_2d(A  + (size_t)m0 * K + k0n, lds_off(Ash[cur ^ 1]), 32, 128,
                        (uint64_t)K, 3, 3);
            tdm_load_2d(Wt + (size_t)n0 * K + k0n, lds_off(Bsh[cur ^ 1]), 32, 128,
                        (uint64_t)K, 3, 3);
        }
        gemm_tile_compute(Ash[cur], Bsh[cur], wm, wn, ln, hf, acc);
    }
#else
    const int arow = tid >> 1, ach = tid & 1;     // 128 rows x 2 chunks of 16
    for (int t = 0; t < nk; ++t) {
        const int k0 = t << 5;
        const __bf16* asrc = A + (size_t)(m0 + arow) * K + k0 + ach * 16;
        *(uint4*)(Ash[0] + arow * ALD + ach * 16)     = *(const uint4*)asrc;
        *(uint4*)(Ash[0] + arow * ALD + ach * 16 + 8) = *(const uint4*)(asrc + 8);
        const __bf16* bsrc = Wt + (size_t)(n0 + arow) * K + k0 + ach * 16;
        *(uint4*)(Bsh[0] + arow * ALD + ach * 16)     = *(const uint4*)bsrc;
        *(uint4*)(Bsh[0] + arow * ALD + ach * 16 + 8) = *(const uint4*)(bsrc + 8);
        __syncthreads();
        gemm_tile_compute(Ash[0], Bsh[0], wm, wn, ln, hf, acc);
        __syncthreads();
    }
#endif

    // epilogue: C lane layout -> row = r + hf*8, col = ln within each 16x16 frag
#pragma unroll
    for (int i = 0; i < 2; ++i)
#pragma unroll
        for (int j = 0; j < 4; ++j) {
            const int col = n0 + wn * 64 + j * 16 + ln;
            const float bv = bias[col];
#pragma unroll
            for (int r = 0; r < 8; ++r) {
                const int row = m0 + wm * 32 + i * 16 + hf * 8 + r;
                float v = acc[i][j][r] + bv;
                const size_t idx = (size_t)row * N + col;
                if (MODE == 0) {
                    ((float*)Cout)[idx] = v;
                } else if (MODE == 1) {
                    ((__bf16*)Cout)[idx] = (__bf16)v;
                } else if (MODE == 2) {
                    ((float*)Cout)[idx] = 1.0f / (1.0f + __expf(-v));
                } else {
                    ((float*)Cout)[idx] += v;
                }
            }
        }
}

// ---------------------------------------------------------------------------
// Flash attention with ALiBi (non-causal |q-k|), WMMA bf16, fused output gate:
// writes gc = sigmoid(gate) * ctx directly as bf16 (input to out-projection).
// grid = (SEQ/128, NHEADS, BATCH), block = 256 (8 waves, 16 q-rows per wave).
// ---------------------------------------------------------------------------
#define KT     64    // keys per tile
#define KSH_LD 136   // K tile [key][d] stride  (TDM: 64-dword rows + 4-dword pad)
#define VT_LD  72    // V^T [d][key] / P [row][key] stride

__global__ __launch_bounds__(256) void flash_attn_kernel(
    const __bf16* __restrict__ qkv, const int* __restrict__ amask,
    const float* __restrict__ slopes, const float* __restrict__ gsig,
    __bf16* __restrict__ gc)
{
    __shared__ __bf16 KV[DHEAD * VT_LD];       // K tile (64x136) or V^T (128x72)
    __shared__ __bf16 Pw[8 * 16 * VT_LD];      // per-wave P tiles (16x64 each)

    const int b = blockIdx.z, hh = blockIdx.y, qt = blockIdx.x;
    const int tid = threadIdx.x;
    const int w = tid >> 5, lane = tid & 31, ln = lane & 15, hf = lane >> 4;
    const float slope = slopes[hh];

    // Q fragments: 16 rows x 128 d, loaded straight from global in A layout
    const int qrow = qt * 128 + w * 16 + ln;
    const __bf16* qp = qkv + (size_t)(b * SEQ + qrow) * QKV_LD + hh * DHEAD;
    v16bf qf[4];
#pragma unroll
    for (int kc = 0; kc < 4; ++kc) qf[kc] = load_frag_a(qp + kc * 32, hf);

    v8f of[8];
#pragma unroll
    for (int j = 0; j < 8; ++j) of[j] = (v8f){};
    float mprev[8], lsum[8];
#pragma unroll
    for (int r = 0; r < 8; ++r) { mprev[r] = -1e30f; lsum[r] = 0.f; }

    for (int t = 0; t < SEQ / KT; ++t) {
        const int kbase = t * KT;
        // ---- stage K tile [KT][DHEAD] (TDM DMA or cooperative fallback)
#if USE_TDM
        if (tid < 32) {
            tdm_load_2d(qkv + (size_t)(b * SEQ + kbase) * QKV_LD + DMODEL + hh * DHEAD,
                        lds_off(KV), DHEAD, KT, (uint64_t)QKV_LD, 5, 3);
            __builtin_amdgcn_s_wait_tensorcnt(0);
        }
        __syncthreads();
#else
#pragma unroll
        for (int it = 0; it < 2; ++it) {
            const int idx = tid + it * 256;          // 0..511
            const int kk = idx >> 3, dc = idx & 7;
            const __bf16* src = qkv + (size_t)(b * SEQ + kbase + kk) * QKV_LD
                                + DMODEL + hh * DHEAD + dc * 16;
            *(uint4*)(KV + kk * KSH_LD + dc * 16)     = *(const uint4*)src;
            *(uint4*)(KV + kk * KSH_LD + dc * 16 + 8) = *(const uint4*)(src + 8);
        }
        __syncthreads();
#endif

        // ---- S = Q @ K^T  (16 x 64 per wave)
        v8f sf[KT / 16];
#pragma unroll
        for (int j = 0; j < KT / 16; ++j) {
            v8f acc = (v8f){};
#pragma unroll
            for (int kc = 0; kc < 4; ++kc) {
                v16bf bb = load_frag_b(KV + (j * 16 + ln) * KSH_LD + kc * 32, hf);
                acc = __builtin_amdgcn_wmma_f32_16x16x32_bf16(
                    false, qf[kc], false, bb, (short)0, acc, false, false);
            }
            sf[j] = acc;
        }

        // ---- scale + alibi + mask; running row max
        float mcur[8];
#pragma unroll
        for (int r = 0; r < 8; ++r) mcur[r] = -1e30f;
#pragma unroll
        for (int j = 0; j < KT / 16; ++j) {
            const int key = kbase + j * 16 + ln;
            const int mk = amask[b * SEQ + key];
#pragma unroll
            for (int r = 0; r < 8; ++r) {
                const int qq = qt * 128 + w * 16 + hf * 8 + r;
                float v = sf[j][r] * INV_SQRT_D - slope * fabsf((float)(qq - key));
                if (mk == 0) v = -1e30f;
                sf[j][r] = v;
                mcur[r] = fmaxf(mcur[r], v);
            }
        }
#pragma unroll
        for (int off = 8; off >= 1; off >>= 1)
#pragma unroll
            for (int r = 0; r < 8; ++r)
                mcur[r] = fmaxf(mcur[r], __shfl_xor(mcur[r], off, 32));

        float cs[8], rs[8];
#pragma unroll
        for (int r = 0; r < 8; ++r) {
            const float mn = fmaxf(mprev[r], mcur[r]);
            cs[r] = __expf(mprev[r] - mn);
            mprev[r] = mn;
            rs[r] = 0.f;
        }
#pragma unroll
        for (int j = 0; j < KT / 16; ++j)
#pragma unroll
            for (int r = 0; r < 8; ++r) {
                const float pv = __expf(sf[j][r] - mprev[r]);
                sf[j][r] = pv;
                rs[r] += pv;
            }
#pragma unroll
        for (int off = 8; off >= 1; off >>= 1)
#pragma unroll
            for (int r = 0; r < 8; ++r) rs[r] += __shfl_xor(rs[r], off, 32);
#pragma unroll
        for (int r = 0; r < 8; ++r) lsum[r] = lsum[r] * cs[r] + rs[r];
#pragma unroll
        for (int j = 0; j < 8; ++j)
#pragma unroll
            for (int r = 0; r < 8; ++r) of[j][r] *= cs[r];

        // ---- spill P (bf16) to per-wave LDS for A-frag relayout
#pragma unroll
        for (int j = 0; j < KT / 16; ++j)
#pragma unroll
            for (int r = 0; r < 8; ++r)
                Pw[(w * 16 + hf * 8 + r) * VT_LD + j * 16 + ln] = (__bf16)sf[j][r];
        __syncthreads();   // all waves done with K tile

        // ---- stage V^T [DHEAD][KT] into the same buffer (manual: transpose)
#pragma unroll
        for (int it = 0; it < 2; ++it) {
            const int idx = tid + it * 256;
            const int kk = idx >> 3, dc = idx & 7;
            const __bf16* src = qkv + (size_t)(b * SEQ + kbase + kk) * QKV_LD
                                + 2 * DMODEL + hh * DHEAD + dc * 16;
            Frag16 tv;
            tv.q[0] = *(const uint4*)src;
            tv.q[1] = *(const uint4*)(src + 8);
#pragma unroll
            for (int j = 0; j < 16; ++j) KV[(dc * 16 + j) * VT_LD + kk] = tv.v[j];
        }
        __syncthreads();

        // ---- O += P @ V
#pragma unroll
        for (int kc = 0; kc < KT / 32; ++kc) {
            const __bf16* pp = Pw + (w * 16 + ln) * VT_LD + kc * 32;
            const v16bf af = load_frag_a(pp, hf);
#pragma unroll
            for (int j = 0; j < 8; ++j) {
                v16bf bb = load_frag_b(KV + (j * 16 + ln) * VT_LD + kc * 32, hf);
                of[j] = __builtin_amdgcn_wmma_f32_16x16x32_bf16(
                    false, af, false, bb, (short)0, of[j], false, false);
            }
        }
        __syncthreads();   // before next K tile overwrites KV
    }

    // ---- normalize, gate, and write gc (bf16) in (b, q, h*DHEAD+d) layout
#pragma unroll
    for (int j = 0; j < 8; ++j)
#pragma unroll
        for (int r = 0; r < 8; ++r) {
            const int row = qt * 128 + w * 16 + hf * 8 + r;
            const size_t idx =
                (size_t)(b * SEQ + row) * DMODEL + hh * DHEAD + j * 16 + ln;
            gc[idx] = (__bf16)(gsig[idx] * (of[j][r] / lsum[r]));
        }
}

// ---------------------------------------------------------------------------
// SwiGLU: h = silu(u1) * u2 from up output (N = 2*DFFN)
// ---------------------------------------------------------------------------
__global__ __launch_bounds__(256) void swiglu_kernel(const __bf16* __restrict__ up,
                                                     __bf16* __restrict__ h, int n) {
    int i = blockIdx.x * 256 + threadIdx.x;
    if (i < n) {
        const int t = i >> 13;            // / DFFN
        const int j = i & (DFFN - 1);
        const float u1 = (float)up[(size_t)t * (2 * DFFN) + j];
        const float u2 = (float)up[(size_t)t * (2 * DFFN) + DFFN + j];
        h[i] = (__bf16)(u1 / (1.0f + __expf(-u1)) * u2);
    }
}

// ---------------------------------------------------------------------------
// Launcher
// ---------------------------------------------------------------------------
extern "C" void kernel_launch(void* const* d_in, const int* in_sizes, int n_in,
                              void* d_out, int out_size, void* d_ws, size_t ws_size,
                              hipStream_t stream) {
    const float* x      = (const float*)d_in[0];
    const int*   amask  = (const int*)  d_in[1];
    const float* rms_w  = (const float*)d_in[2];
    const float* qkv_w  = (const float*)d_in[3];
    const float* qkv_b  = (const float*)d_in[4];
    const float* up_w   = (const float*)d_in[5];
    const float* up_b   = (const float*)d_in[6];
    const float* dw_w   = (const float*)d_in[7];
    const float* dw_b   = (const float*)d_in[8];
    const float* gate_w = (const float*)d_in[9];
    const float* gate_b = (const float*)d_in[10];
    const float* out_w  = (const float*)d_in[11];
    const float* out_b  = (const float*)d_in[12];
    const float* slopes = (const float*)d_in[13];
    float* out = (float*)d_out;

    char* p = (char*)d_ws;
    auto alloc = [&](size_t elems, size_t esz) -> void* {
        void* r = (void*)p;
        p += (elems * esz + 255) & ~(size_t)255;
        return r;
    };
    __bf16* xnbf  = (__bf16*)alloc((size_t)NTOK * DMODEL, 2);
    __bf16* qkvwt = (__bf16*)alloc((size_t)DMODEL * QKV_LD, 2);      // (6144 x 2048)
    __bf16* gatewt= (__bf16*)alloc((size_t)DMODEL * DMODEL, 2);
    __bf16* outwt = (__bf16*)alloc((size_t)DMODEL * DMODEL, 2);
    __bf16* upwt  = (__bf16*)alloc((size_t)DMODEL * 2 * DFFN, 2);    // (16384 x 2048)
    __bf16* dwwt  = (__bf16*)alloc((size_t)DFFN * DMODEL, 2);        // (2048 x 8192)
    __bf16* qkvo  = (__bf16*)alloc((size_t)NTOK * QKV_LD, 2);
    float*  gsig  = (float*) alloc((size_t)NTOK * DMODEL, 4);
    __bf16* gcbf  = (__bf16*)alloc((size_t)NTOK * DMODEL, 2);
    __bf16* upo   = (__bf16*)alloc((size_t)NTOK * 2 * DFFN, 2);
    __bf16* hbf   = (__bf16*)alloc((size_t)NTOK * DFFN, 2);

    auto tcast = [&](const float* s, __bf16* d, int K, int N) {
        transpose_cast_kernel<<<dim3(N / 32, K / 32), 256, 0, stream>>>(s, d, K, N);
    };
    // 1) weights -> transposed bf16 (Wt is N x K so GEMM B-tiles are row-contiguous)
    tcast(qkv_w,  qkvwt,  DMODEL, QKV_LD);
    tcast(gate_w, gatewt, DMODEL, DMODEL);
    tcast(out_w,  outwt,  DMODEL, DMODEL);
    tcast(up_w,   upwt,   DMODEL, 2 * DFFN);
    tcast(dw_w,   dwwt,   DFFN,   DMODEL);

    // 2) RMSNorm
    rmsnorm_kernel<<<NTOK, 256, 0, stream>>>(x, rms_w, xnbf);

    // 3) QKV projection (bf16 out for attention)
    gemm_bf16_kernel<1><<<dim3(QKV_LD / 128, NTOK / 128), 256, 0, stream>>>(
        xnbf, qkvwt, qkv_b, qkvo, NTOK, QKV_LD, DMODEL);

    // 4) gate projection + sigmoid
    gemm_bf16_kernel<2><<<dim3(DMODEL / 128, NTOK / 128), 256, 0, stream>>>(
        xnbf, gatewt, gate_b, gsig, NTOK, DMODEL, DMODEL);

    // 5) flash attention with ALiBi, fused sigmoid-gate multiply -> gcbf
    flash_attn_kernel<<<dim3(SEQ / 128, NHEADS, BATCH), 256, 0, stream>>>(
        qkvo, amask, slopes, gsig, gcbf);

    // 6) FFN up projection
    gemm_bf16_kernel<1><<<dim3(2 * DFFN / 128, NTOK / 128), 256, 0, stream>>>(
        xnbf, upwt, up_b, upo, NTOK, 2 * DFFN, DMODEL);

    // 7) SwiGLU
    swiglu_kernel<<<(int)(((size_t)NTOK * DFFN + 255) / 256), 256, 0, stream>>>(
        upo, hbf, NTOK * DFFN);

    // 8) FFN down projection -> d_out (= ffn_out, includes dw_b)
    gemm_bf16_kernel<0><<<dim3(DMODEL / 128, NTOK / 128), 256, 0, stream>>>(
        hbf, dwwt, dw_b, out, NTOK, DMODEL, DFFN);

    // 9) out projection, accumulated into d_out (final = attn_out + ffn_out)
    gemm_bf16_kernel<3><<<dim3(DMODEL / 128, NTOK / 128), 256, 0, stream>>>(
        gcbf, outwt, out_b, out, NTOK, DMODEL, DMODEL);
}
